// MultiHeadAttention_2551210574110
// MI455X (gfx1250) — compile-verified, matrix operand order checked by reading, not run on hardware
//
#include <hip/hip_runtime.h>

typedef __attribute__((ext_vector_type(16))) __bf16 v16bf;
typedef __attribute__((ext_vector_type(8)))  __bf16 v8bf;   // 16 bytes
typedef __attribute__((ext_vector_type(8)))  float  v8f;
typedef __attribute__((ext_vector_type(4)))  int    v4i;

constexpr int Bn  = 2;
constexpr int Tn  = 2048;
constexpr int Cn  = 2048;
constexpr int Hn  = 16;
constexpr int HSn = 128;
constexpr int Mn  = Bn * Tn;   // 4096 token rows

// ---------------------------------------------------------------------------
// Async global->LDS 16B copy (CDNA5 GLOBAL_LOAD_ASYNC_TO_LDS_B128) with
// safe fallback when the builtin is not declared by this toolchain.
// Probe-learned signature: (v4i AS1* gaddr, v4i AS3* lds, imm offset, imm cpol)
// ---------------------------------------------------------------------------
#if __has_builtin(__builtin_amdgcn_global_load_async_to_lds_b128)
#define HAVE_ASYNC_LDS 1
typedef __attribute__((address_space(1))) v4i as1_v4i;
typedef __attribute__((address_space(3))) v4i as3_v4i;
#define CP16(g, l)                                                             \
    __builtin_amdgcn_global_load_async_to_lds_b128(                            \
        (as1_v4i*)(unsigned long long)(g), (as3_v4i*)(l), 0, 0)
#else
#define CP16(g, l) (*(v8bf*)(l) = *(const v8bf*)(g))
#endif

static __device__ __forceinline__ void async_wait() {
#if defined(HAVE_ASYNC_LDS)
#if __has_builtin(__builtin_amdgcn_s_wait_asynccnt)
    __builtin_amdgcn_s_wait_asynccnt(0);
#else
    asm volatile("s_wait_asynccnt 0" ::: "memory");
#endif
#endif
}

// Fragment loader: elements [0..7] at p[0..7], [8..15] at p[16..23]
// (both halves 16B aligned in all our LDS layouts) -> two ds_load_b128.
static __device__ __forceinline__ v16bf load_frag(const __bf16* p) {
    v8bf lo = *(const v8bf*)p;
    v8bf hi = *(const v8bf*)(p + 16);
    return __builtin_shufflevector(lo, hi, 0, 1, 2, 3, 4, 5, 6, 7,
                                           8, 9, 10, 11, 12, 13, 14, 15);
}

// ---------------------------------------------------------------------------
// fp32 -> bf16 conversion (grid-stride)
// ---------------------------------------------------------------------------
__global__ void cvt_f32_bf16(const float* __restrict__ in, __bf16* __restrict__ out, int n) {
    int i = blockIdx.x * blockDim.x + threadIdx.x;
    int stride = gridDim.x * blockDim.x;
    for (; i < n; i += stride) out[i] = (__bf16)in[i];
}

// ---------------------------------------------------------------------------
// C[M,N] = A[M,K]bf16 * B[K,N]bf16 + bias.f32
// outMode 0: bf16 -> outBf ; outMode 1: f32 -> outF
// block = 64x64 tile, 256 threads (8 waves), wave -> 16x32 slab
// ---------------------------------------------------------------------------
__global__ __launch_bounds__(256)
void gemm_bf16(const __bf16* __restrict__ A, const __bf16* __restrict__ Bm,
               const float* __restrict__ bias, __bf16* __restrict__ outBf,
               float* __restrict__ outF, int Ndim, int Kdim, int outMode) {
    __shared__ __bf16 As[64][40];    // [m][k], 80B rows (16B aligned)
    __shared__ __bf16 BsT[64][40];   // [n][k] transposed, 80B rows

    const int tid  = threadIdx.x;
    const int lane = tid & 31;
    const int wave = tid >> 5;
    const int wr   = wave & 3;       // rows 16*wr .. +15
    const int wc   = wave >> 2;      // cols 32*wc .. +31
    const int m0   = blockIdx.y * 64;
    const int n0   = blockIdx.x * 64;
    const int hm   = lane & 15;
    const int half = lane >> 4;

    v8f acc0 = {}, acc1 = {};

    for (int k0 = 0; k0 < Kdim; k0 += 32) {
        __syncthreads();
        {   // A tile: 64x32, one async b128 per thread (straight copy)
            int r = tid >> 2, seg = tid & 3;
            const __bf16* g = A + (size_t)(m0 + r) * Kdim + k0 + seg * 8;
            CP16(g, &As[r][seg * 8]);
            // B tile: 32x64 -> transpose into BsT[n][k]
            int r2 = tid >> 3, s2 = tid & 7;
            const __bf16* g2 = Bm + (size_t)(k0 + r2) * Ndim + n0 + s2 * 8;
            v8bf bv = *(const v8bf*)g2;
            #pragma unroll
            for (int e = 0; e < 8; ++e) BsT[s2 * 8 + e][r2] = bv[e];
            async_wait();
        }
        __syncthreads();

        v16bf af  = load_frag(&As[wr * 16 + hm][half * 8]);
        v16bf bf0 = load_frag(&BsT[wc * 32 + hm][half * 8]);
        v16bf bf1 = load_frag(&BsT[wc * 32 + 16 + hm][half * 8]);

        acc0 = __builtin_amdgcn_wmma_f32_16x16x32_bf16(false, af, false, bf0, (short)0, acc0, false, false);
        acc1 = __builtin_amdgcn_wmma_f32_16x16x32_bf16(false, af, false, bf1, (short)0, acc1, false, false);
    }

    // epilogue: bias + store
    const int col0 = n0 + wc * 32 + hm;
    const int col1 = col0 + 16;
    const float b0 = bias ? bias[col0] : 0.f;
    const float b1 = bias ? bias[col1] : 0.f;
    #pragma unroll
    for (int r = 0; r < 8; ++r) {
        int row = m0 + wr * 16 + r + half * 8;
        float v0 = acc0[r] + b0;
        float v1 = acc1[r] + b1;
        if (outMode == 0) {
            outBf[(size_t)row * Ndim + col0] = (__bf16)v0;
            outBf[(size_t)row * Ndim + col1] = (__bf16)v1;
        } else {
            outF[(size_t)row * Ndim + col0] = v0;
            outF[(size_t)row * Ndim + col1] = v1;
        }
    }
}

// ---------------------------------------------------------------------------
// Causal flash attention. grid = (T/64, H, B), block = 128 (4 waves).
// Wave w handles query rows q0 = 64*blockIdx.x + 16*w. hs = 128.
// Q/K/V/O layout: [B*T, C] bf16, head h occupies cols h*128..h*128+127.
// ---------------------------------------------------------------------------
__global__ __launch_bounds__(128)
void flash_attn(const __bf16* __restrict__ Q, const __bf16* __restrict__ K,
                const __bf16* __restrict__ V, __bf16* __restrict__ O) {
    __shared__ __bf16 Ks[32][136];     // [key][dim], 272B rows (16B aligned)
    __shared__ __bf16 VsT[128][40];    // [dim][key] transposed, 80B rows
    __shared__ __bf16 Ps[4][16][40];   // per-wave P staging, 80B rows

    const int tid  = threadIdx.x;
    const int lane = tid & 31;
    const int wave = tid >> 5;
    const int hm   = lane & 15;
    const int half = lane >> 4;
    const int qt   = blockIdx.x;
    const int h    = blockIdx.y;
    const int b    = blockIdx.z;
    const int q0   = qt * 64 + wave * 16;

    // Q fragments: 16x128 per wave as 4 A-frags, held in registers
    v16bf qf[4];
    {
        const size_t rowQ = (size_t)(b * Tn + q0 + hm) * Cn + h * HSn;
        #pragma unroll
        for (int f = 0; f < 4; ++f) qf[f] = load_frag(Q + rowQ + f * 32 + half * 8);
    }

    float mi[8], li[8];
    v8f   o[8];
    #pragma unroll
    for (int r = 0; r < 8; ++r) { mi[r] = -3.0e38f; li[r] = 0.f; }
    #pragma unroll
    for (int d = 0; d < 8; ++d) o[d] = (v8f){};

    const float scale = 0.08838834764831845f;   // 1/sqrt(128)
    const int nkt = (qt + 1) * 2;               // causal bound, key tiles of 32

    for (int kt = 0; kt < nkt; ++kt) {
        __syncthreads();
        {   // K tile: straight copy, 4 async b128 per thread
            int krow = tid >> 2, seg = tid & 3;
            const size_t gbase = (size_t)(b * Tn + kt * 32 + krow) * Cn + h * HSn + seg * 32;
            const __bf16* gK = K + gbase;
            #pragma unroll
            for (int j = 0; j < 4; ++j) CP16(gK + j * 8, &Ks[krow][seg * 32 + j * 8]);
            // V tile: transpose into VsT[dim][key]
            const __bf16* gV = V + gbase;
            #pragma unroll
            for (int j = 0; j < 4; ++j) {
                v8bf vv = *(const v8bf*)(gV + j * 8);
                #pragma unroll
                for (int e = 0; e < 8; ++e) VsT[seg * 32 + j * 8 + e][krow] = vv[e];
            }
            async_wait();
        }
        __syncthreads();

        // S = Q * K^T : 16x32 scores per wave (two 16x16 accs), K-dim 128
        v8f s0 = {}, s1 = {};
        #pragma unroll
        for (int f = 0; f < 4; ++f) {
            v16bf kf0 = load_frag(&Ks[hm][f * 32 + half * 8]);        // key = hm
            v16bf kf1 = load_frag(&Ks[16 + hm][f * 32 + half * 8]);   // key = 16+hm
            s0 = __builtin_amdgcn_wmma_f32_16x16x32_bf16(false, qf[f], false, kf0, (short)0, s0, false, false);
            s1 = __builtin_amdgcn_wmma_f32_16x16x32_bf16(false, qf[f], false, kf1, (short)0, s1, false, false);
        }

        // online softmax per row (C-layout: lane=col, slot r -> row r+8*half)
        #pragma unroll
        for (int r = 0; r < 8; ++r) {
            const int qrow = q0 + r + 8 * half;
            const int c0 = kt * 32 + hm;
            const int c1 = c0 + 16;
            float a0 = s0[r] * scale; if (c0 > qrow) a0 = -3.0e38f;
            float a1 = s1[r] * scale; if (c1 > qrow) a1 = -3.0e38f;
            float mloc = fmaxf(a0, a1);
            #pragma unroll
            for (int d2 = 1; d2 < 16; d2 <<= 1) mloc = fmaxf(mloc, __shfl_xor(mloc, d2, 32));
            float mnew = fmaxf(mi[r], mloc);
            float p0 = __expf(a0 - mnew);
            float p1 = __expf(a1 - mnew);
            float ls = p0 + p1;
            #pragma unroll
            for (int d2 = 1; d2 < 16; d2 <<= 1) ls += __shfl_xor(ls, d2, 32);
            float alpha = __expf(mi[r] - mnew);
            li[r] = li[r] * alpha + ls;
            mi[r] = mnew;
            #pragma unroll
            for (int d = 0; d < 8; ++d) o[d][r] *= alpha;
            Ps[wave][r + 8 * half][hm]      = (__bf16)p0;
            Ps[wave][r + 8 * half][16 + hm] = (__bf16)p1;
        }
        // P staging is per-wave LDS: only this wave's DS ops must complete
        asm volatile("s_wait_dscnt 0" ::: "memory");

        // P (16x32) as A-frag; O += P * V over eight 16-col output tiles
        v16bf pf = load_frag(&Ps[wave][hm][half * 8]);
        #pragma unroll
        for (int d = 0; d < 8; ++d) {
            v16bf vf = load_frag(&VsT[d * 16 + hm][half * 8]);
            o[d] = __builtin_amdgcn_wmma_f32_16x16x32_bf16(false, pf, false, vf, (short)0, o[d], false, false);
        }
    }

    // normalize + store bf16
    #pragma unroll
    for (int r = 0; r < 8; ++r) {
        const float inv = 1.0f / li[r];
        const size_t row = (size_t)(b * Tn + q0 + r + 8 * half);
        #pragma unroll
        for (int d = 0; d < 8; ++d)
            O[row * Cn + h * HSn + d * 16 + hm] = (__bf16)(o[d][r] * inv);
    }
}

// ---------------------------------------------------------------------------
extern "C" void kernel_launch(void* const* d_in, const int* in_sizes, int n_in,
                              void* d_out, int out_size, void* d_ws, size_t ws_size,
                              hipStream_t stream) {
    (void)in_sizes; (void)n_in; (void)out_size; (void)ws_size;
    const float* idx = (const float*)d_in[0];
    const float* Wq  = (const float*)d_in[1];
    const float* bq  = (const float*)d_in[2];
    const float* Wk  = (const float*)d_in[3];
    const float* bk  = (const float*)d_in[4];
    const float* Wv  = (const float*)d_in[5];
    const float* bv  = (const float*)d_in[6];
    const float* Wo  = (const float*)d_in[7];
    const float* bo  = (const float*)d_in[8];

    char* ws = (char*)d_ws;
    size_t off = 0;
    auto alloc = [&](size_t bytes) -> void* {
        void* p = ws + off;
        off += (bytes + 255) & ~size_t(255);
        return p;
    };
    __bf16* xb  = (__bf16*)alloc((size_t)Mn * Cn * 2);
    __bf16* wqb = (__bf16*)alloc((size_t)Cn * Cn * 2);
    __bf16* wkb = (__bf16*)alloc((size_t)Cn * Cn * 2);
    __bf16* wvb = (__bf16*)alloc((size_t)Cn * Cn * 2);
    __bf16* wob = (__bf16*)alloc((size_t)Cn * Cn * 2);
    __bf16* qb  = (__bf16*)alloc((size_t)Mn * Cn * 2);
    __bf16* kb  = (__bf16*)alloc((size_t)Mn * Cn * 2);
    __bf16* vb  = (__bf16*)alloc((size_t)Mn * Cn * 2);
    __bf16* ab  = (__bf16*)alloc((size_t)Mn * Cn * 2);

    const int nIdx = Mn * Cn;
    const int nW   = Cn * Cn;
    cvt_f32_bf16<<<4096, 256, 0, stream>>>(idx, xb,  nIdx);
    cvt_f32_bf16<<<4096, 256, 0, stream>>>(Wq,  wqb, nW);
    cvt_f32_bf16<<<4096, 256, 0, stream>>>(Wk,  wkb, nW);
    cvt_f32_bf16<<<4096, 256, 0, stream>>>(Wv,  wvb, nW);
    cvt_f32_bf16<<<4096, 256, 0, stream>>>(Wo,  wob, nW);

    dim3 gg(Cn / 64, Mn / 64);
    gemm_bf16<<<gg, 256, 0, stream>>>(xb, wqb, bq, qb, nullptr, Cn, Cn, 0);
    gemm_bf16<<<gg, 256, 0, stream>>>(xb, wkb, bk, kb, nullptr, Cn, Cn, 0);
    gemm_bf16<<<gg, 256, 0, stream>>>(xb, wvb, bv, vb, nullptr, Cn, Cn, 0);

    dim3 ga(Tn / 64, Hn, Bn);
    flash_attn<<<ga, 128, 0, stream>>>(qb, kb, vb, ab);

    gemm_bf16<<<gg, 256, 0, stream>>>(ab, wob, bo, nullptr, (float*)d_out, Cn, Cn, 1);
}
